// Dynamic_GCN_FusionLayer_89043261980845
// MI455X (gfx1250) — matrix operand order repaired and checked by reading, not compiled, not run on hardware
//
#include <hip/hip_runtime.h>

#define BATCH 16
#define NNODE 2048
#define DIM   256

typedef __attribute__((ext_vector_type(16))) __bf16 v16bf;
typedef __attribute__((ext_vector_type(8)))  __bf16 v8bf;
typedef __attribute__((ext_vector_type(8)))  float  v8f;

static __device__ __forceinline__ v8f wmma_bf16(v16bf a, v16bf b, v8f c) {
  // 8 args: (neg_a, A, neg_b, B, c_mod, C, reuse_a, reuse_b)
  return __builtin_amdgcn_wmma_f32_16x16x32_bf16(false, a, false, b, (short)0, c,
                                                 false, false);
}

// A operand: 16x32 bf16 tile, row-major source, rows contiguous (stride lda elems).
// Lane (0-15): row=lane, K chunks {kg..kg+7} in v[0..7], {kg+16..kg+23} in v[8..15],
// kg = (lane>=16)?8:0  (per ISA 7.12.2 16-bit A layout).
static __device__ __forceinline__ v16bf load_a_frag(const __bf16* base, int lda) {
  const int lane = threadIdx.x & 31;
  const int row  = lane & 15;
  const int kg   = (lane >> 4) << 3;           // 0 or 8
  const __bf16* p = base + (size_t)row * lda + kg;
  v8bf lo = *(const v8bf*)(p);
  v8bf hi = *(const v8bf*)(p + 16);
  v16bf r;
#pragma unroll
  for (int i = 0; i < 8; ++i) { r[i] = lo[i]; r[i + 8] = hi[i]; }
  return r;
}

// B operand: 32x16 bf16 tile (KxN) taken from TRANSPOSED storage Bt[N][K]
// (stride ldb). Lane col = lane&15, K chunks {kg..kg+7} in v[0..7], {kg+16..kg+23} in v[8..15],
// kg = (lane>=16)?8:0  (matches A layout).
static __device__ __forceinline__ v16bf load_bt_frag(const __bf16* baseT, int ldb) {
  const int lane = threadIdx.x & 31;
  const int col  = lane & 15;
  const int kg   = (lane >> 4) << 3;           // 0 or 8
  const __bf16* p = baseT + (size_t)col * ldb + kg;
  v8bf lo = *(const v8bf*)(p);
  v8bf hi = *(const v8bf*)(p + 16);
  v16bf r;
#pragma unroll
  for (int i = 0; i < 8; ++i) { r[i] = lo[i]; r[i + 8] = hi[i]; }
  return r;
}

// ---------------- elementwise converts ----------------
__global__ void convert_x_kernel(const float* __restrict__ x, __bf16* __restrict__ xb, int n) {
  int i = blockIdx.x * 256 + threadIdx.x;
  if (i < n) xb[i] = (__bf16)x[i];
}

// Wt[w][n][k] = W_w[k][n]  (transpose + convert), w in {gv, eq, ek}
__global__ void convert_w_kernel(const float* __restrict__ Wgv,
                                 const float* __restrict__ Weq,
                                 const float* __restrict__ Wek,
                                 __bf16* __restrict__ Wt) {
  int i = blockIdx.x * 256 + threadIdx.x;          // 3*256*256
  int w = i >> 16, rem = i & 65535;
  int n = rem >> 8, k = rem & 255;
  const float* src = (w == 0) ? Wgv : (w == 1) ? Weq : Wek;
  Wt[i] = (__bf16)src[k * 256 + n];
}

// ---------------- tabular path (tiny, fp32) ----------------
// Per batch: T_q, T_X (-> d_out), u = Wgk @ T_q, c = T_q . bgk
__global__ void tab_kernel(const float* __restrict__ tab,
                           const float* __restrict__ Wtq, const float* __restrict__ btq,
                           const float* __restrict__ Wtv, const float* __restrict__ btv,
                           const float* __restrict__ Wgk, const float* __restrict__ bgk,
                           float* __restrict__ Tq, float* __restrict__ u,
                           float* __restrict__ c, float* __restrict__ out_tx) {
  int b = blockIdx.x, j = threadIdx.x;
  __shared__ float tabL[DIM];
  __shared__ float TqL[DIM];
  tabL[j] = tab[b * DIM + j];
  __syncthreads();
  float sq = btq[j], sv = btv[j];
  for (int i = 0; i < DIM; ++i) {
    float t = tabL[i];
    sq += t * Wtq[i * DIM + j];
    sv += t * Wtv[i * DIM + j];
  }
  Tq[b * DIM + j] = sq;
  TqL[j] = sq;
  out_tx[b * DIM + j] = (sv >= 0.f) ? sv : 0.01f * sv;   // leaky_relu(T_X)
  __syncthreads();
  float su = 0.f;
  for (int h = 0; h < DIM; ++h) su += Wgk[j * DIM + h] * TqL[h];
  u[b * DIM + j] = su;
  if (j == 0) {
    float cc = 0.f;
    for (int h = 0; h < DIM; ++h) cc += TqL[h] * bgk[h];
    c[b] = cc;
  }
}

// ---------------- fused node linears (WMMA) ----------------
// w=0: Gv = leaky(X Wgv + bgv) -> f32 ; w=1: q = relu(...) bf16 ; w=2: k bf16
__global__ void linear_kernel(const __bf16* __restrict__ Xb,
                              const __bf16* __restrict__ Wt,
                              const float* __restrict__ bgv,
                              const float* __restrict__ beq,
                              const float* __restrict__ bek,
                              float* __restrict__ Gv,
                              __bf16* __restrict__ qb,
                              __bf16* __restrict__ kb) {
  const int b = blockIdx.z / 3, w = blockIdx.z % 3;
  const int wave = threadIdx.x >> 5;
  const int m0 = (blockIdx.x * 8 + wave) * 16;
  const int n0 = blockIdx.y * 16;
  const __bf16* Xbase = Xb + (size_t)b * NNODE * DIM + (size_t)m0 * DIM;
  const __bf16* Wbase = Wt + (size_t)w * DIM * DIM + (size_t)n0 * DIM;
  v8f acc = {};
#pragma unroll
  for (int kk = 0; kk < DIM / 32; ++kk) {
    v16bf af = load_a_frag(Xbase + kk * 32, DIM);
    v16bf bf = load_bt_frag(Wbase + kk * 32, DIM);
    acc = wmma_bf16(af, bf, acc);
  }
  const int lane = threadIdx.x & 31;
  const int col = lane & 15, mb = (lane >> 4) << 3;
  const int n = n0 + col;
  const float* bias = (w == 0) ? bgv : (w == 1) ? beq : bek;
  const float bn = bias[n];
#pragma unroll
  for (int r = 0; r < 8; ++r) {
    int gm = m0 + mb + r;
    size_t idx = ((size_t)b * NNODE + gm) * DIM + n;
    float v = acc[r] + bn;
    if (w == 0) {
      Gv[idx] = (v >= 0.f) ? v : 0.01f * v;
    } else {
      v = fmaxf(v, 0.f);
      if (w == 1) qb[idx] = (__bf16)v; else kb[idx] = (__bf16)v;
    }
  }
}

// ---------------- node_att logits: s[b,y] = X[y].u[b] + c[b] ----------------
__global__ void score_kernel(const float* __restrict__ X, const float* __restrict__ u,
                             const float* __restrict__ c, float* __restrict__ s) {
  int b = blockIdx.y;
  int row = blockIdx.x * 8 + (threadIdx.x >> 5);
  int lane = threadIdx.x & 31;
  const float* xr = X + ((size_t)b * NNODE + row) * DIM;
  const float* uu = u + b * DIM;
  float sum = 0.f;
  for (int h = lane; h < DIM; h += 32) sum += xr[h] * uu[h];
#pragma unroll
  for (int off = 16; off > 0; off >>= 1) sum += __shfl_xor(sum, off, 32);
  if (lane == 0) s[b * NNODE + row] = sum + c[b];
}

// ---------------- softmax over 2048 + add 1 ----------------
__global__ void softmax_kernel(const float* __restrict__ s, float* __restrict__ a1) {
  int b = blockIdx.x, tid = threadIdx.x;
  __shared__ float red[256];
  const float* sb = s + (size_t)b * NNODE;
  float v[8];
#pragma unroll
  for (int k = 0; k < 8; ++k) v[k] = sb[k * 256 + tid];
  float mx = v[0];
#pragma unroll
  for (int k = 1; k < 8; ++k) mx = fmaxf(mx, v[k]);
  red[tid] = mx; __syncthreads();
  for (int st = 128; st > 0; st >>= 1) {
    if (tid < st) red[tid] = fmaxf(red[tid], red[tid + st]);
    __syncthreads();
  }
  mx = red[0]; __syncthreads();
  float e[8], se = 0.f;
#pragma unroll
  for (int k = 0; k < 8; ++k) { e[k] = __expf(v[k] - mx); se += e[k]; }
  red[tid] = se; __syncthreads();
  for (int st = 128; st > 0; st >>= 1) {
    if (tid < st) red[tid] += red[tid + st];
    __syncthreads();
  }
  float inv = 1.f / red[0];
#pragma unroll
  for (int k = 0; k < 8; ++k) a1[(size_t)b * NNODE + k * 256 + tid] = 1.f + e[k] * inv;
}

// ------- attention + adj mask + self loop -> Ahat (bf16, WMMA, 32x32/wave) ---
// Wave computes P[x0..x0+31, y0..y0+31]: t1[a][b2] = q_{x0+16a} . k_{y0+16b2}^T
// and t2[a][b2] = k_{x0+16a} . q_{y0+16b2}^T = att^T tile. 8 WMMA per K-step
// against 8 fragment loads (register-level operand reuse).
__global__ void att_kernel(const __bf16* __restrict__ qb, const __bf16* __restrict__ kb,
                           const float* __restrict__ adj, __bf16* __restrict__ Ahat) {
  const int b = blockIdx.z;
  const int x0 = blockIdx.x * 32;
  const int y0 = (blockIdx.y * 8 + (threadIdx.x >> 5)) * 32;
  const __bf16* qbb = qb + (size_t)b * NNODE * DIM;
  const __bf16* kbb = kb + (size_t)b * NNODE * DIM;
  v8f t1[2][2] = {{{}, {}}, {{}, {}}};
  v8f t2[2][2] = {{{}, {}}, {{}, {}}};
#pragma unroll
  for (int kk = 0; kk < DIM / 32; ++kk) {
    const int k0 = kk * 32;
    v16bf aq[2], ak[2], bk[2], bq[2];
#pragma unroll
    for (int a = 0; a < 2; ++a) {
      aq[a] = load_a_frag(qbb + (size_t)(x0 + 16 * a) * DIM + k0, DIM);
      ak[a] = load_a_frag(kbb + (size_t)(x0 + 16 * a) * DIM + k0, DIM);
      bk[a] = load_bt_frag(kbb + (size_t)(y0 + 16 * a) * DIM + k0, DIM);
      bq[a] = load_bt_frag(qbb + (size_t)(y0 + 16 * a) * DIM + k0, DIM);
    }
#pragma unroll
    for (int a = 0; a < 2; ++a)
#pragma unroll
      for (int b2 = 0; b2 < 2; ++b2) {
        t1[a][b2] = wmma_bf16(aq[a], bk[b2], t1[a][b2]);   // att[x,y]
        t2[a][b2] = wmma_bf16(ak[a], bq[b2], t2[a][b2]);   // att[y,x]
      }
  }
  const int lane = threadIdx.x & 31;
  const int col = lane & 15, mb = (lane >> 4) << 3;
  const float scale = 0.0625f;                        // 256^-0.5
#pragma unroll
  for (int a = 0; a < 2; ++a)
#pragma unroll
    for (int b2 = 0; b2 < 2; ++b2) {
      const int gn = y0 + 16 * b2 + col;
#pragma unroll
      for (int r = 0; r < 8; ++r) {
        int gm = x0 + 16 * a + mb + r;
        float z = scale * (t1[a][b2][r] + t2[a][b2][r]);
        float P = 1.f / (1.f + __expf(-z));
        float ad = adj[((size_t)b * NNODE + gm) * NNODE + gn];
        float ah = ad * P + ((gm == gn) ? 1.f : 0.f);   // adj*att + I
        Ahat[((size_t)b * NNODE + gm) * NNODE + gn] = (__bf16)ah;
      }
    }
}

// ---------------- deterministic row sums -> rsd = rsqrt(max(d,1e-12)) -------
__global__ void rowsum_kernel(const __bf16* __restrict__ Ahat, float* __restrict__ rsd) {
  int b = blockIdx.y;
  int row = blockIdx.x * 8 + (threadIdx.x >> 5);
  int lane = threadIdx.x & 31;
  const __bf16* ar = Ahat + ((size_t)b * NNODE + row) * NNODE;
  float s = 0.f;
  for (int y = lane; y < NNODE; y += 32) s += (float)ar[y];
#pragma unroll
  for (int off = 16; off > 0; off >>= 1) s += __shfl_xor(s, off, 32);
  if (lane == 0) rsd[(size_t)b * NNODE + row] = rsqrtf(fmaxf(s, 1e-12f));
}

// ---- H[d][y] = bf16( rsd[y] * (1+att[y]) * Gv[y][d] )  (LDS-tiled transpose)
__global__ void scale_transpose_kernel(const float* __restrict__ Gv,
                                       const float* __restrict__ a1,
                                       const float* __restrict__ rsd,
                                       __bf16* __restrict__ Ht) {
  __shared__ float t[32][33];
  const int b = blockIdx.z;
  const int y0 = blockIdx.x * 32, n0 = blockIdx.y * 32;
  const int tx = threadIdx.x & 31, ty = threadIdx.x >> 5;   // 32 x 8
#pragma unroll
  for (int r = 0; r < 4; ++r) {
    int yy = ty + r * 8;
    t[yy][tx] = Gv[((size_t)b * NNODE + y0 + yy) * DIM + n0 + tx];
  }
  __syncthreads();
  const int gy = y0 + tx;
  const float scl = rsd[(size_t)b * NNODE + gy] * a1[(size_t)b * NNODE + gy];
#pragma unroll
  for (int r = 0; r < 4; ++r) {
    int nn = ty + r * 8;
    Ht[(size_t)b * DIM * NNODE + (size_t)(n0 + nn) * NNODE + gy] =
        (__bf16)(t[tx][nn] * scl);
  }
}

// ------- aggregation: out = diag(rsd) Ahat H  (WMMA, 32x32 per wave) ---------
// K = 2048; per K-step: 2 A-fragments + 2 B-fragments -> 4 WMMA.
__global__ void agg_kernel(const __bf16* __restrict__ Ahat, const __bf16* __restrict__ Ht,
                           const float* __restrict__ rsd, float* __restrict__ out) {
  const int b = blockIdx.z;
  const int m0 = blockIdx.x * 32;
  const int n0 = (threadIdx.x >> 5) * 32;             // 8 waves cover N=256
  const __bf16* Ab = Ahat + (size_t)b * NNODE * NNODE + (size_t)m0 * NNODE;
  const __bf16* Hb = Ht + (size_t)b * DIM * NNODE + (size_t)n0 * NNODE;
  v8f acc[2][2] = {{{}, {}}, {{}, {}}};
  for (int kk = 0; kk < NNODE / 32; ++kk) {
    const int k0 = kk * 32;
    v16bf af[2], bf[2];
#pragma unroll
    for (int a = 0; a < 2; ++a) {
      af[a] = load_a_frag(Ab + (size_t)(16 * a) * NNODE + k0, NNODE);
      bf[a] = load_bt_frag(Hb + (size_t)(16 * a) * NNODE + k0, NNODE);
    }
#pragma unroll
    for (int a = 0; a < 2; ++a)
#pragma unroll
      for (int b2 = 0; b2 < 2; ++b2)
        acc[a][b2] = wmma_bf16(af[a], bf[b2], acc[a][b2]);
  }
  const int lane = threadIdx.x & 31;
  const int col = lane & 15, mb = (lane >> 4) << 3;
#pragma unroll
  for (int a = 0; a < 2; ++a)
#pragma unroll
    for (int b2 = 0; b2 < 2; ++b2) {
#pragma unroll
      for (int r = 0; r < 8; ++r) {
        int gm = m0 + 16 * a + mb + r;
        out[((size_t)b * NNODE + gm) * DIM + n0 + 16 * b2 + col] =
            acc[a][b2][r] * rsd[(size_t)b * NNODE + gm];
      }
    }
}

extern "C" void kernel_launch(void* const* d_in, const int* in_sizes, int n_in,
                              void* d_out, int out_size, void* d_ws, size_t ws_size,
                              hipStream_t stream) {
  const float* tab  = (const float*)d_in[0];
  const float* node = (const float*)d_in[1];
  const float* adj  = (const float*)d_in[2];
  const float* Wgv  = (const float*)d_in[3];
  const float* bgv  = (const float*)d_in[4];
  const float* Wgk  = (const float*)d_in[5];
  const float* bgk  = (const float*)d_in[6];
  const float* Weq  = (const float*)d_in[7];
  const float* beq  = (const float*)d_in[8];
  const float* Wek  = (const float*)d_in[9];
  const float* bek  = (const float*)d_in[10];
  const float* Wtq  = (const float*)d_in[11];
  const float* btq  = (const float*)d_in[12];
  const float* Wtv  = (const float*)d_in[13];
  const float* btv  = (const float*)d_in[14];
  float* out = (float*)d_out;                   // [T_X(16*256), G_X(16*2048*256)]
  float* out_gx = out + BATCH * DIM;

  char* ws = (char*)d_ws;
  size_t o = 0;
  auto take = [&](size_t bytes) { size_t r = o; o += (bytes + 255) & ~(size_t)255; return r; };
  __bf16* Xb   = (__bf16*)(ws + take((size_t)BATCH * NNODE * DIM * 2));   // 16 MB
  __bf16* Wt   = (__bf16*)(ws + take((size_t)3 * DIM * DIM * 2));
  __bf16* qb   = (__bf16*)(ws + take((size_t)BATCH * NNODE * DIM * 2));   // 16 MB
  __bf16* kb   = (__bf16*)(ws + take((size_t)BATCH * NNODE * DIM * 2));   // 16 MB
  float*  Gv   = (float*) (ws + take((size_t)BATCH * NNODE * DIM * 4));   // 32 MB
  __bf16* Ht   = (__bf16*)(ws + take((size_t)BATCH * DIM * NNODE * 2));   // 16 MB
  __bf16* Ahat = (__bf16*)(ws + take((size_t)BATCH * NNODE * NNODE * 2)); // 128 MB
  float*  Tq   = (float*) (ws + take((size_t)BATCH * DIM * 4));
  float*  u    = (float*) (ws + take((size_t)BATCH * DIM * 4));
  float*  c    = (float*) (ws + take((size_t)BATCH * 4));
  float*  s    = (float*) (ws + take((size_t)BATCH * NNODE * 4));
  float*  a1   = (float*) (ws + take((size_t)BATCH * NNODE * 4));
  float*  rsd  = (float*) (ws + take((size_t)BATCH * NNODE * 4));

  const int nx = BATCH * NNODE * DIM;
  convert_x_kernel<<<(nx + 255) / 256, 256, 0, stream>>>(node, Xb, nx);
  convert_w_kernel<<<(3 * DIM * DIM) / 256, 256, 0, stream>>>(Wgv, Weq, Wek, Wt);
  tab_kernel<<<BATCH, 256, 0, stream>>>(tab, Wtq, btq, Wtv, btv, Wgk, bgk, Tq, u, c, out);
  linear_kernel<<<dim3(16, 16, BATCH * 3), 256, 0, stream>>>(Xb, Wt, bgv, beq, bek, Gv, qb, kb);
  score_kernel<<<dim3(NNODE / 8, BATCH), 256, 0, stream>>>(node, u, c, s);
  softmax_kernel<<<BATCH, 256, 0, stream>>>(s, a1);
  att_kernel<<<dim3(NNODE / 32, 8, BATCH), 256, 0, stream>>>(qb, kb, adj, Ahat);
  rowsum_kernel<<<dim3(NNODE / 8, BATCH), 256, 0, stream>>>(Ahat, rsd);
  scale_transpose_kernel<<<dim3(NNODE / 32, DIM / 32, BATCH), 256, 0, stream>>>(Gv, a1, rsd, Ht);
  agg_kernel<<<dim3(NNODE / 32, 1, BATCH), 256, 0, stream>>>(Ahat, Ht, rsd, out_gx);
}
